// VectorQuantizer_34969623724288
// MI455X (gfx1250) — matrix operand order repaired and checked by reading, not compiled, not run on hardware
//
#include <hip/hip_runtime.h>
#include <hip/hip_fp16.h>

typedef __attribute__((ext_vector_type(16))) _Float16 v16h;
typedef __attribute__((ext_vector_type(8)))  float    v8f;

#define DDIM    256
#define TCODES  128
#define RSTRIDE (DDIM + 16)   // halfs; 544B rows -> 32B aligned, staggered banks
#define DECAY   0.99f
#define BETA    0.25f

__device__ __forceinline__ unsigned ordf(float f) {
  unsigned u = __float_as_uint(f);
  return (u & 0x80000000u) ? ~u : (u | 0x80000000u);
}

// ---------------- init scratch ----------------
__global__ void k_init(unsigned long long* __restrict__ colbest,
                       unsigned* __restrict__ hist, int K) {
  int i = blockIdx.x * 256 + threadIdx.x;
  if (i < K) { colbest[i] = 0ull; hist[i] = 0u; }
}

// ------- L2-normalize codebook rows -> f16 -------
__global__ __launch_bounds__(256) void k_norm_embed(const float* __restrict__ emb,
                                                    _Float16* __restrict__ en, int K) {
  int wave = threadIdx.x >> 5, lane = threadIdx.x & 31;
  int row = blockIdx.x * 8 + wave;
  if (row >= K) return;
  const float* er = emb + (size_t)row * DDIM;
  float4 f0 = *(const float4*)(er + lane * 8);
  float4 f1 = *(const float4*)(er + lane * 8 + 4);
  float ss = f0.x*f0.x + f0.y*f0.y + f0.z*f0.z + f0.w*f0.w
           + f1.x*f1.x + f1.y*f1.y + f1.z*f1.z + f1.w*f1.w;
  #pragma unroll
  for (int m = 16; m >= 1; m >>= 1) ss += __shfl_xor(ss, m, 32);
  float rinv = 1.0f / fmaxf(sqrtf(ss), 1e-12f);
  _Float16* o = en + (size_t)row * DDIM + lane * 8;
  o[0] = (_Float16)(f0.x * rinv); o[1] = (_Float16)(f0.y * rinv);
  o[2] = (_Float16)(f0.z * rinv); o[3] = (_Float16)(f0.w * rinv);
  o[4] = (_Float16)(f1.x * rinv); o[5] = (_Float16)(f1.y * rinv);
  o[6] = (_Float16)(f1.z * rinv); o[7] = (_Float16)(f1.w * rinv);
}

// ------- fused GEMM (WMMA) + row argmax + per-code max -------
__global__ __launch_bounds__(256) void k_sim(const float* __restrict__ z,
                                             const _Float16* __restrict__ en,
                                             int* __restrict__ idx,
                                             unsigned long long* __restrict__ colbest,
                                             int K) {
  __shared__ _Float16 sB[TCODES * RSTRIDE];
  __shared__ unsigned long long sCol[TCODES];

  const int tid  = threadIdx.x;
  const int wave = tid >> 5, lane = tid & 31;
  const int seg  = lane >> 4, r = lane & 15;
  const int rowbase = blockIdx.x * 128 + wave * 16;
  const int row = rowbase + r;

  // Load 16 z rows into WMMA A fragments; lane(seg=0) holds K = c*32+0..15,
  // lane(seg=1) holds K = c*32+16..31  (contiguous K order, matching B).
  v16h A[8];
  float ss = 0.f;
  const float* zr = z + (size_t)row * DDIM;
  #pragma unroll
  for (int c = 0; c < 8; ++c) {
    const int b0 = c * 32 + seg * 16;
    float4 f0 = *(const float4*)(zr + b0);
    float4 f1 = *(const float4*)(zr + b0 + 4);
    float4 f2 = *(const float4*)(zr + b0 + 8);
    float4 f3 = *(const float4*)(zr + b0 + 12);
    ss += f0.x*f0.x + f0.y*f0.y + f0.z*f0.z + f0.w*f0.w;
    ss += f1.x*f1.x + f1.y*f1.y + f1.z*f1.z + f1.w*f1.w;
    ss += f2.x*f2.x + f2.y*f2.y + f2.z*f2.z + f2.w*f2.w;
    ss += f3.x*f3.x + f3.y*f3.y + f3.z*f3.z + f3.w*f3.w;
    A[c][0]  = (_Float16)f0.x; A[c][1]  = (_Float16)f0.y;
    A[c][2]  = (_Float16)f0.z; A[c][3]  = (_Float16)f0.w;
    A[c][4]  = (_Float16)f1.x; A[c][5]  = (_Float16)f1.y;
    A[c][6]  = (_Float16)f1.z; A[c][7]  = (_Float16)f1.w;
    A[c][8]  = (_Float16)f2.x; A[c][9]  = (_Float16)f2.y;
    A[c][10] = (_Float16)f2.z; A[c][11] = (_Float16)f2.w;
    A[c][12] = (_Float16)f3.x; A[c][13] = (_Float16)f3.y;
    A[c][14] = (_Float16)f3.z; A[c][15] = (_Float16)f3.w;
  }
  ss += __shfl_xor(ss, 16, 32);              // lanes L, L+16 hold row L&15
  float rinv = 1.0f / fmaxf(sqrtf(ss), 1e-12f);
  _Float16 hr = (_Float16)rinv;
  #pragma unroll
  for (int c = 0; c < 8; ++c) {
    #pragma unroll
    for (int j = 0; j < 16; ++j) A[c][j] *= hr;
  }

  float bestv[8];
  int   bestc[8];
  #pragma unroll
  for (int v = 0; v < 8; ++v) { bestv[v] = -1e30f; bestc[v] = 0; }

  const int nch = K / TCODES;
  for (int ch = 0; ch < nch; ++ch) {
    if (tid < TCODES) sCol[tid] = 0ull;
    // stage 128 normalized codes (f16) into LDS
    const _Float16* src = en + (size_t)ch * TCODES * DDIM;
    for (int i = tid; i < TCODES * (DDIM / 8); i += 256) {
      int cr = i >> 5;            // code row (32 chunks of 8 halfs per row)
      int cc = (i & 31) * 8;
      *(uint4*)(&sB[cr * RSTRIDE + cc]) = *(const uint4*)(src + (size_t)cr * DDIM + cc);
    }
    __syncthreads();

    #pragma unroll
    for (int t = 0; t < TCODES / 16; ++t) {
      v8f acc = {};
      // B fragment: lane = column n = code t*16+r; halfs = 16 contiguous K (seg*16 offset)
      const _Float16* bp = &sB[(t * 16 + r) * RSTRIDE + seg * 16];
      #pragma unroll
      for (int c = 0; c < 8; ++c) {
        v16h Bv = *(const v16h*)(bp + c * 32);
        acc = __builtin_amdgcn_wmma_f32_16x16x32_f16(false, A[c], false, Bv,
                                                     (short)0, acc, false, false);
      }
      const int code = ch * TCODES + t * 16 + r;
      // running per-row argmax (lane sees codes == r (mod 16), ascending -> '>' keeps first)
      #pragma unroll
      for (int v = 0; v < 8; ++v)
        if (acc[v] > bestv[v]) { bestv[v] = acc[v]; bestc[v] = code; }
      // per-code max over this wave's 16 rows (acc[v] is row rowbase+seg*8+v)
      float cm = acc[0]; int cv = 0;
      #pragma unroll
      for (int v = 1; v < 8; ++v)
        if (acc[v] > cm) { cm = acc[v]; cv = v; }
      int grow = rowbase + seg * 8 + cv;
      float om  = __shfl_xor(cm, 16, 32);
      int ogrow = __shfl_xor(grow, 16, 32);
      if (om > cm || (om == cm && ogrow < grow)) { cm = om; grow = ogrow; }
      if (seg == 0) {
        unsigned long long key =
            ((unsigned long long)ordf(cm) << 32) | (unsigned)(~(unsigned)grow);
        atomicMax(&sCol[t * 16 + r], key);
      }
    }
    __syncthreads();
    if (tid < TCODES) atomicMax(&colbest[ch * TCODES + tid], sCol[tid]);
  }

  // butterfly over the 16 lanes that share rows (same seg half)
  #pragma unroll
  for (int m = 1; m < 16; m <<= 1) {
    #pragma unroll
    for (int v = 0; v < 8; ++v) {
      float ov = __shfl_xor(bestv[v], m, 32);
      int   oc = __shfl_xor(bestc[v], m, 32);
      if (ov > bestv[v] || (ov == bestv[v] && oc < bestc[v])) {
        bestv[v] = ov; bestc[v] = oc;
      }
    }
  }
  if (r == 0) {
    #pragma unroll
    for (int v = 0; v < 8; ++v) idx[rowbase + seg * 8 + v] = bestc[v];
  }
}

// ------- gather z_q, histogram, loss partials -------
__global__ __launch_bounds__(256) void k_gather(const float* __restrict__ z,
                                                const float* __restrict__ emb,
                                                const int* __restrict__ idx,
                                                float* __restrict__ zq,
                                                float* __restrict__ partials,
                                                unsigned* __restrict__ hist) {
  int wave = threadIdx.x >> 5, lane = threadIdx.x & 31;
  int row = blockIdx.x * 8 + wave;
  int k = idx[row];
  const float* er = emb + (size_t)k * DDIM;
  const float* zr = z + (size_t)row * DDIM;
  float4 e0 = *(const float4*)(er + lane * 8);
  float4 e1 = *(const float4*)(er + lane * 8 + 4);
  float4 z0 = *(const float4*)(zr + lane * 8);
  float4 z1 = *(const float4*)(zr + lane * 8 + 4);
  *(float4*)(zq + (size_t)row * DDIM + lane * 8)     = e0;
  *(float4*)(zq + (size_t)row * DDIM + lane * 8 + 4) = e1;
  float s = 0.f, d;
  d = e0.x - z0.x; s += d * d;  d = e0.y - z0.y; s += d * d;
  d = e0.z - z0.z; s += d * d;  d = e0.w - z0.w; s += d * d;
  d = e1.x - z1.x; s += d * d;  d = e1.y - z1.y; s += d * d;
  d = e1.z - z1.z; s += d * d;  d = e1.w - z1.w; s += d * d;
  if (lane == 0) atomicAdd(&hist[k], 1u);
  #pragma unroll
  for (int m = 16; m >= 1; m >>= 1) s += __shfl_xor(s, m, 32);
  __shared__ float sw[8];
  if (lane == 0) sw[wave] = s;
  __syncthreads();
  if (threadIdx.x == 0) {
    float t = 0.f;
    #pragma unroll
    for (int i = 0; i < 8; ++i) t += sw[i];
    partials[blockIdx.x] = t;
  }
}

// ------- final loss reduction -------
__global__ void k_loss(const float* __restrict__ partials, int P,
                       float* __restrict__ loss_out, float scale) {
  int lane = threadIdx.x & 31, wave = threadIdx.x >> 5;
  float s = 0.f;
  for (int i = threadIdx.x; i < P; i += 256) s += partials[i];
  #pragma unroll
  for (int m = 16; m >= 1; m >>= 1) s += __shfl_xor(s, m, 32);
  __shared__ float sw[8];
  if (lane == 0) sw[wave] = s;
  __syncthreads();
  if (threadIdx.x == 0) {
    float t = 0.f;
    #pragma unroll
    for (int i = 0; i < 8; ++i) t += sw[i];
    *loss_out = (1.0f + BETA) * t * scale;   // beta*MSE + MSE (stop_grads keep value)
  }
}

// ------- EMA probs + codebook reinit blend -------
__global__ __launch_bounds__(256) void k_final(const float* __restrict__ z,
                                               const float* __restrict__ emb,
                                               const float* __restrict__ prob,
                                               const unsigned* __restrict__ hist,
                                               const unsigned long long* __restrict__ colbest,
                                               float* __restrict__ out_emb,
                                               float* __restrict__ out_prob,
                                               int K, int N) {
  int wave = threadIdx.x >> 5, lane = threadIdx.x & 31;
  int code = blockIdx.x * 8 + wave;
  if (code >= K) return;
  unsigned crow = ~(unsigned)(colbest[code] & 0xFFFFFFFFull);
  float np = prob[code] * DECAY + ((float)hist[code] / (float)N) * (1.0f - DECAY);
  float re = expf(-(np * (float)K * 10.0f) / (1.0f - DECAY) - 0.001f);
  const float* zr = z + (size_t)crow * DDIM;
  const float* er = emb + (size_t)code * DDIM;
  float4 za = *(const float4*)(zr + lane * 8);
  float4 zb = *(const float4*)(zr + lane * 8 + 4);
  float4 ea = *(const float4*)(er + lane * 8);
  float4 eb = *(const float4*)(er + lane * 8 + 4);
  float om = 1.0f - re;
  float4 na, nb;
  na.x = ea.x * om + za.x * re; na.y = ea.y * om + za.y * re;
  na.z = ea.z * om + za.z * re; na.w = ea.w * om + za.w * re;
  nb.x = eb.x * om + zb.x * re; nb.y = eb.y * om + zb.y * re;
  nb.z = eb.z * om + zb.z * re; nb.w = eb.w * om + zb.w * re;
  *(float4*)(out_emb + (size_t)code * DDIM + lane * 8)     = na;
  *(float4*)(out_emb + (size_t)code * DDIM + lane * 8 + 4) = nb;
  if (lane == 0) out_prob[code] = np;
}

extern "C" void kernel_launch(void* const* d_in, const int* in_sizes, int n_in,
                              void* d_out, int out_size, void* d_ws, size_t ws_size,
                              hipStream_t stream) {
  (void)n_in; (void)out_size; (void)ws_size;
  const float* z    = (const float*)d_in[0];
  const float* emb  = (const float*)d_in[1];
  const float* prob = (const float*)d_in[2];
  const int K = in_sizes[2];              // 2048
  const int D = in_sizes[1] / K;          // 256
  const int N = in_sizes[0] / D;          // 131072

  char* ws = (char*)d_ws;
  size_t o = 0;
  _Float16* en = (_Float16*)(ws + o);               o += (size_t)K * DDIM * 2;
  unsigned long long* colbest =
      (unsigned long long*)(ws + o);                o += (size_t)K * 8;
  int* idx = (int*)(ws + o);                        o += (size_t)N * 4;
  unsigned* hist = (unsigned*)(ws + o);             o += (size_t)K * 4;
  float* partials = (float*)(ws + o);               o += (size_t)(N / 8) * 4;

  float* zq       = (float*)d_out;
  float* loss_out = zq + (size_t)N * DDIM;
  float* out_emb  = loss_out + 1;
  float* out_prob = out_emb + (size_t)K * DDIM;

  k_init<<<(K + 255) / 256, 256, 0, stream>>>(colbest, hist, K);
  k_norm_embed<<<K / 8, 256, 0, stream>>>(emb, en, K);
  k_sim<<<N / 128, 256, 0, stream>>>(z, en, idx, colbest, K);
  k_gather<<<N / 8, 256, 0, stream>>>(z, emb, idx, zq, partials, hist);
  k_loss<<<1, 256, 0, stream>>>(partials, N / 8, loss_out,
                                1.0f / ((float)N * (float)DDIM));
  k_final<<<K / 8, 256, 0, stream>>>(z, emb, prob, hist, colbest,
                                     out_emb, out_prob, K, N);
}
